// Model_88407606821209
// MI455X (gfx1250) — matrix operand order repaired and checked by reading, not compiled, not run on hardware
//
#include <hip/hip_runtime.h>

// ---------------------------------------------------------------------------
// GCN encoder stack for MI455X (gfx1250, wave32).
// GEMMs run on v_wmma_f32_16x16x32_bf16 (bf16 A/B, f32 accumulate), with each
// wave computing a 16x64 output strip (4 N-tiles) so one A fragment feeds 4
// WMMAs (2.5 b128 loads per WMMA instead of 4).
// Edge aggregation is the bandwidth-bound part (~4.8 GB -> ~0.2ms @ 23.3TB/s);
// the 100MB accumulator is L2-resident (192MB L2) so f32 atomics are cheap.
// ---------------------------------------------------------------------------

typedef __bf16 bf16_t;
typedef bf16_t v16bf __attribute__((ext_vector_type(16)));
typedef bf16_t v8bf  __attribute__((ext_vector_type(8)));
typedef float  v8f   __attribute__((ext_vector_type(8)));

static __device__ __forceinline__ bf16_t f2bf(float f) {
  union { float f; unsigned u; } v; v.f = f;
  unsigned u = v.u;
  unsigned r = u + 0x7FFFu + ((u >> 16) & 1u);   // round-to-nearest-even
  unsigned short h = (unsigned short)(r >> 16);
  bf16_t o;
  __builtin_memcpy(&o, &h, sizeof(h));
  return o;
}

// ---------------- degree / normalization ----------------------------------
__global__ void k_fill_f32(float* __restrict__ p, int n, float v) {
  int i = blockIdx.x * blockDim.x + threadIdx.x;
  if (i < n) p[i] = v;
}

__global__ void k_count_dst(const int* __restrict__ dst, float* __restrict__ deg, int E) {
  int i = blockIdx.x * blockDim.x + threadIdx.x;
  if (i < E) atomicAdd(&deg[dst[i]], 1.0f);
}

__global__ void k_rsqrt(const float* __restrict__ deg, float* __restrict__ inv, int n) {
  int i = blockIdx.x * blockDim.x + threadIdx.x;
  if (i < n) inv[i] = rsqrtf(deg[i]);
}

// ---------------- casts / weight packing ----------------------------------
__global__ void k_cast_bf16(const float* __restrict__ src, bf16_t* __restrict__ dst, long long n) {
  long long i = (long long)blockIdx.x * blockDim.x + threadIdx.x;
  if (i < n) dst[i] = f2bf(src[i]);
}

// W: [K, Nc] f32 row-major  ->  Wt: [Nc, K] bf16 row-major (so a WMMA B-column
// is a contiguous K-run: one 32B load per lane per k-step).
__global__ void k_pack_w(const float* __restrict__ W, bf16_t* __restrict__ Wt, int K, int Nc) {
  int i = blockIdx.x * blockDim.x + threadIdx.x;
  if (i >= K * Nc) return;
  int k = i / Nc, n = i - k * Nc;
  Wt[(long long)n * K + k] = f2bf(W[i]);
}

// ---------------- WMMA GEMM: C[M,Nc] = A[M,K] x B[K,Nc] -------------------
// A: bf16 row-major, Bt: bf16 [Nc,K] row-major, C: f32 row-major.
// One wave per 16x64 output strip (4 N-tiles sharing one A fragment).
// M multiple of 16, Nc multiple of 64, K multiple of 32.
__global__ __launch_bounds__(256) void k_gemm_bf16(
    const bf16_t* __restrict__ A, const bf16_t* __restrict__ Bt,
    float* __restrict__ C, int M, int K, int Nc)
{
  const int NT = 4;                            // N-tiles per wave
  const int wavesPerBlock = blockDim.x >> 5;
  int wid = blockIdx.x * wavesPerBlock + (threadIdx.x >> 5);
  int ngroups = Nc >> 6;                       // strips of 64 columns
  int mtiles = M >> 4;
  if (wid >= mtiles * ngroups) return;         // whole-wave uniform exit
  int tm = wid / ngroups;
  int tg = wid - tm * ngroups;                 // strip index; columns tg*64..+63

  int lane = threadIdx.x & 31;
  int lr = lane & 15;                          // A row / B,C column within tile
  int hi = lane >> 4;                          // lane half select

  // A fragment (16x32 bf16): lanes 0-15 hold K 0..15, lanes 16-31 hold
  // K 16..31 for their row -> two contiguous 16B chunks per lane.
  const bf16_t* arow = A  + (long long)(tm * 16 + lr) * K + hi * 16;
  // B fragment (32x16 bf16): lane holds column, K-run (hi*16 .. +15)
  // -> one contiguous 32B chunk per lane (Bt row-major in K).
  const bf16_t* brow = Bt + (long long)(tg * 64 + lr) * K + hi * 16;
  const long long bstep = (long long)16 * K;   // next 16-column tile of Bt

  v8f c0 = {}, c1 = {}, c2 = {}, c3 = {};
  for (int k0 = 0; k0 < K; k0 += 32) {
    __builtin_prefetch(arow + k0 + 128, 0, 1);   // global_prefetch_b8, A stream
    v8bf alo = *(const v8bf*)(arow + k0);
    v8bf ahi = *(const v8bf*)(arow + k0 + 8);
    v16bf a;
#pragma unroll
    for (int i = 0; i < 8; ++i) { a[i] = alo[i]; a[i + 8] = ahi[i]; }
    v16bf b0 = *(const v16bf*)(brow + k0);
    v16bf b1 = *(const v16bf*)(brow + bstep + k0);
    v16bf b2 = *(const v16bf*)(brow + 2 * bstep + k0);
    v16bf b3 = *(const v16bf*)(brow + 3 * bstep + k0);
    c0 = __builtin_amdgcn_wmma_f32_16x16x32_bf16(false, a, false, b0, (short)0, c0, false, false);
    c1 = __builtin_amdgcn_wmma_f32_16x16x32_bf16(false, a, false, b1, (short)0, c1, false, false);
    c2 = __builtin_amdgcn_wmma_f32_16x16x32_bf16(false, a, false, b2, (short)0, c2, false, false);
    c3 = __builtin_amdgcn_wmma_f32_16x16x32_bf16(false, a, false, b3, (short)0, c3, false, false);
  }

  // C/D layout: VGPR r -> row r (lanes 0-15) / row r+8 (lanes 16-31), col = lr.
  float* cp = C + (long long)(tm * 16 + hi * 8) * Nc + tg * 64 + lr;
#pragma unroll
  for (int r = 0; r < 8; ++r) {
    long long ro = (long long)r * Nc;
    cp[ro]      = c0[r];
    cp[ro + 16] = c1[r];
    cp[ro + 32] = c2[r];
    cp[ro + 48] = c3[r];
  }
}

// ---------------- aggregation ----------------------------------------------
// agg[v,f] = h[v,f] * inv[v]^2 + bias[f]   (self-loop term + bias)
__global__ void k_agg_init(const float* __restrict__ h, const float* __restrict__ inv,
                           const float* __restrict__ bias, float* __restrict__ agg,
                           long long n, int d) {
  long long i = (long long)blockIdx.x * blockDim.x + threadIdx.x;
  if (i >= n) return;
  int v = (int)(i / d);
  int f = (int)(i - (long long)v * d);
  float s = inv[v];
  agg[i] = h[i] * s * s + bias[f];
}

// one block per edge: coalesced gather of h[src,:], coalesced f32 atomics into
// agg[dst,:] (accumulator is L2-resident).
__global__ __launch_bounds__(256) void k_agg_edges(
    const float* __restrict__ h, const int* __restrict__ src, const int* __restrict__ dst,
    const float* __restrict__ inv, float* __restrict__ agg, int E, int d) {
  int e = blockIdx.x;
  if (e >= E) return;
  int s = src[e], t = dst[e];
  float coef = inv[s] * inv[t];
  const float* hs = h + (long long)s * d;
  float* at = agg + (long long)t * d;
  for (int f = threadIdx.x; f < d; f += blockDim.x)
    atomicAdd(&at[f], hs[f] * coef);
}

// ReLU finalize; optionally emits f32 (model output) and/or bf16 (next layer A).
__global__ void k_finalize(const float* __restrict__ agg, float* __restrict__ outf,
                           bf16_t* __restrict__ outb, long long n) {
  long long i = (long long)blockIdx.x * blockDim.x + threadIdx.x;
  if (i >= n) return;
  float v = agg[i];
  v = v > 0.0f ? v : 0.0f;
  if (outf) outf[i] = v;
  if (outb) outb[i] = f2bf(v);
}

// ---------------------------------------------------------------------------
#define CDIV(a, b) (((a) + (b) - 1) / (b))

extern "C" void kernel_launch(void* const* d_in, const int* in_sizes, int n_in,
                              void* d_out, int out_size, void* d_ws, size_t ws_size,
                              hipStream_t stream) {
  const float* x  = (const float*)d_in[0];
  const int*   e1 = (const int*)d_in[1];
  const int*   e2 = (const int*)d_in[2];
  const float* W1 = (const float*)d_in[3];
  const float* b1 = (const float*)d_in[4];
  const float* W2 = (const float*)d_in[5];
  const float* b2 = (const float*)d_in[6];
  const float* W3 = (const float*)d_in[7];
  const float* b3 = (const float*)d_in[8];
  const float* W4 = (const float*)d_in[9];
  const float* b4 = (const float*)d_in[10];

  const int DIN = 512, DHID = 512, DOUT = 256;
  const int N = in_sizes[0] / DIN;
  const int E = in_sizes[1] / 2;
  const int *src1 = e1, *dst1 = e1 + E;
  const int *src2 = e2, *dst2 = e2 + E;

  // ---- carve workspace ----
  char* ws = (char*)d_ws;
  size_t off = 0;
  auto carve = [&](size_t bytes) -> void* {
    void* p = ws + off;
    off += (bytes + 255) & ~(size_t)255;
    return p;
  };
  float*  inv1   = (float*)carve((size_t)N * 4);
  float*  inv2   = (float*)carve((size_t)N * 4);
  float*  hbuf   = (float*)carve((size_t)N * DHID * 4);   // GEMM out / deg tmp
  float*  aggbuf = (float*)carve((size_t)N * DHID * 4);
  bf16_t* actb   = (bf16_t*)carve((size_t)N * DHID * 2);  // bf16 activations
  bf16_t* wtb    = (bf16_t*)carve((size_t)512 * 512 * 2); // packed weights
  (void)ws_size; (void)n_in; (void)out_size;

  const int T = 256;
  auto degrees = [&](const int* dst, float* inv) {
    k_fill_f32<<<CDIV(N, T), T, 0, stream>>>(hbuf, N, 1.0f);
    k_count_dst<<<CDIV(E, T), T, 0, stream>>>(dst, hbuf, E);
    k_rsqrt<<<CDIV(N, T), T, 0, stream>>>(hbuf, inv, N);
  };
  degrees(dst1, inv1);
  degrees(dst2, inv2);

  auto gemm = [&](const bf16_t* A, const bf16_t* Bt, float* C, int K, int Nc) {
    int strips = (N >> 4) * (Nc >> 6);   // 16x64 strips, 1 wave each
    k_gemm_bf16<<<CDIV(strips, 8), 256, 0, stream>>>(A, Bt, C, N, K, Nc);
  };
  auto layer = [&](const float* W, const float* bias, int K, int Nc,
                   const int* src, const int* dst, const float* inv,
                   float* outf, bf16_t* outb) {
    k_pack_w<<<CDIV(K * Nc, T), T, 0, stream>>>(W, wtb, K, Nc);
    gemm(actb, wtb, hbuf, K, Nc);
    long long n = (long long)N * Nc;
    k_agg_init<<<(unsigned)CDIV(n, T), T, 0, stream>>>(hbuf, inv, bias, aggbuf, n, Nc);
    k_agg_edges<<<E, 256, 0, stream>>>(hbuf, src, dst, inv, aggbuf, E, Nc);
    k_finalize<<<(unsigned)CDIV(n, T), T, 0, stream>>>(aggbuf, outf, outb, n);
  };

  float* z_out  = (float*)d_out;                 // [N, DOUT]
  float* zg_out = (float*)d_out + (size_t)N * DOUT;

  // Layer 1: x -> a1 (bf16 only)
  k_cast_bf16<<<(unsigned)CDIV((long long)N * DIN, T), T, 0, stream>>>(x, actb, (long long)N * DIN);
  layer(W1, b1, DIN,  DHID, src1, dst1, inv1, nullptr, actb);
  // Layer 2: a1 -> z (f32 to d_out + bf16 for layer 3)
  layer(W2, b2, DHID, DOUT, src1, dst1, inv1, z_out,   actb);
  // Layer 3: z -> a3 (bf16 only)
  layer(W3, b3, DOUT, DHID, src2, dst2, inv2, nullptr, actb);
  // Layer 4: a3 -> z_g (f32 to d_out only)
  layer(W4, b4, DHID, DOUT, src2, dst2, inv2, zg_out,  nullptr);
}
